// Codebook_30159260353213
// MI455X (gfx1250) — compile-verified, matrix operand order checked by reading, not run on hardware
//
#include <hip/hip_runtime.h>
#include <math.h>

typedef __attribute__((ext_vector_type(16))) _Float16 v16h;
typedef __attribute__((ext_vector_type(8)))  float    v8f;

#define CB_K   8192      // codebook entries
#define DIM    256       // embedding dim
#define NPOS   16384     // 16*32*32 spatial positions
#define HW     1024      // 32*32
#define NB     16        // batch
#define BETA   0.25f
#define ZQ_ELEMS (NB * DIM * HW)   // 4,194,304

// ---------------------------------------------------------------------------
// Kernel 1: L2-normalize codebook rows -> f16 [K, 256], one wave per row.
// ---------------------------------------------------------------------------
__global__ void k_norm_codebook(const float* __restrict__ cb,
                                _Float16* __restrict__ e_norm) {
    int row  = (blockIdx.x * blockDim.x + threadIdx.x) >> 5;
    int lane = threadIdx.x & 31;
    if (row >= CB_K) return;
    const float* src = cb + (size_t)row * DIM;
    float v[8]; float s = 0.f;
#pragma unroll
    for (int k = 0; k < 8; ++k) { float x = src[lane + 32 * k]; v[k] = x; s += x * x; }
#pragma unroll
    for (int off = 16; off >= 1; off >>= 1) s += __shfl_xor(s, off, 32);
    float scale = 1.0f / fmaxf(sqrtf(s), 1e-12f);
    _Float16* dst = e_norm + (size_t)row * DIM;
#pragma unroll
    for (int k = 0; k < 8; ++k) dst[lane + 32 * k] = (_Float16)(v[k] * scale);
}

// ---------------------------------------------------------------------------
// Kernel 2: L2-normalize z vectors -> f16 [N, 256], one wave per position.
// 32 waves/block cover 32 consecutive hw so the 4KB-strided lines get full
// reuse inside the WGP caches.
// ---------------------------------------------------------------------------
__global__ void k_norm_z(const float* __restrict__ z,
                         _Float16* __restrict__ z_norm) {
    int n    = blockIdx.x * (blockDim.x >> 5) + (threadIdx.x >> 5);
    int lane = threadIdx.x & 31;
    if (n >= NPOS) return;
    int b = n >> 10, hw = n & (HW - 1);
    const float* base = z + ((size_t)b * DIM) * HW + hw;
    float v[8]; float s = 0.f;
#pragma unroll
    for (int k = 0; k < 8; ++k) {
        float x = base[(size_t)(lane + 32 * k) * HW];
        v[k] = x; s += x * x;
    }
#pragma unroll
    for (int off = 16; off >= 1; off >>= 1) s += __shfl_xor(s, off, 32);
    float scale = 1.0f / fmaxf(sqrtf(s), 1e-12f);
    _Float16* dst = z_norm + (size_t)n * DIM;
#pragma unroll
    for (int k = 0; k < 8; ++k) dst[lane + 32 * k] = (_Float16)(v[k] * scale);
}

// ---------------------------------------------------------------------------
// Kernel 3: WMMA similarity + argmax with double-buffered B operands.
// __launch_bounds__(256, 1) lifts the default per-wave VGPR budget so the
// two 64-VGPR B buffers stay in registers (round-2 spilled them to scratch).
// While the 8-WMMA chain consumes tile t, the 16 global_load_b128 for tile
// t+1 are in flight -> loads overlap the matrix pipe instead of gating it.
// ---------------------------------------------------------------------------
__global__ void __launch_bounds__(256, 1)
k_argmax(const _Float16* __restrict__ z_norm,
         const _Float16* __restrict__ e_norm,
         int* __restrict__ idx_ws,
         float* __restrict__ idx_out) {
    const int lane = threadIdx.x & 31;
    const int l15  = lane & 15;
    const int hs   = lane >> 4;           // which 16-half of the K split this lane holds
    const int wave = threadIdx.x >> 5;
    const int rowBase = (blockIdx.x * 8 + wave) * 16;

    // Preload A: 16 rows x 256 dims. Lane l15 = row, hs selects half-chunk.
    v16h a[8];
#pragma unroll
    for (int c = 0; c < 8; ++c) {
        const _Float16* p = z_norm + (size_t)(rowBase + l15) * DIM + c * 32 + hs * 16;
        a[c] = *(const v16h*)p;
    }

    float best[8];
    int   bidx[8];
#pragma unroll
    for (int v = 0; v < 8; ++v) { best[v] = -1e30f; bidx[v] = 0; }

    auto loadB = [&](int cbBase, v16h (&dst)[8]) {
        const _Float16* nb = e_norm + (size_t)(cbBase + l15) * DIM + hs * 16;
#pragma unroll
        for (int c = 0; c < 8; ++c) dst[c] = *(const v16h*)(nb + c * 32);
    };
    auto procTile = [&](int cbBase, const v16h (&bb)[8]) {
        v8f acc = {};
#pragma unroll
        for (int c = 0; c < 8; ++c)
            acc = __builtin_amdgcn_wmma_f32_16x16x32_f16(
                false, a[c], false, bb[c], (short)0, acc, false, false);
        const int col = cbBase + l15;
#pragma unroll
        for (int v = 0; v < 8; ++v)
            if (acc[v] > best[v]) { best[v] = acc[v]; bidx[v] = col; }
    };

    v16h b0[8], b1[8];
    loadB(0, b0);
    for (int t = 0; t < CB_K / 16; t += 2) {
        const int base = t * 16;
        loadB(base + 16, b1);                     // tile t+1 in flight
        if (t + 3 < CB_K / 16)                    // warm L2 for tile t+3
            __builtin_prefetch(e_norm + (size_t)(base + 48 + l15) * DIM, 0, 1);
        procTile(base, b0);                       // consume tile t
        if (t + 2 < CB_K / 16) loadB(base + 32, b0);  // tile t+2 in flight
        procTile(base + 16, b1);                  // consume tile t+1
    }

    // Reduce across the 16 lanes holding the same row set; lowest index wins ties.
#pragma unroll
    for (int off = 8; off >= 1; off >>= 1) {
#pragma unroll
        for (int v = 0; v < 8; ++v) {
            float ov = __shfl_xor(best[v], off, 16);
            int   oi = __shfl_xor(bidx[v], off, 16);
            if (ov > best[v] || (ov == best[v] && oi < bidx[v])) {
                best[v] = ov; bidx[v] = oi;
            }
        }
    }

    if (l15 == 0) {
#pragma unroll
        for (int v = 0; v < 8; ++v) {
            int r = rowBase + hs * 8 + v;
            idx_ws[r]  = bidx[v];
            idx_out[r] = (float)bidx[v];
        }
    }
}

// ---------------------------------------------------------------------------
// Kernel 4a: zero the loss accumulator.
// ---------------------------------------------------------------------------
__global__ void k_zero(float* acc) { *acc = 0.0f; }

// ---------------------------------------------------------------------------
// Kernel 4b: gather + normalize in f32 (exact), write z_q_out in [b,d,h,w]
// layout, accumulate sum((z_q - z)^2). One wave per position.
// ---------------------------------------------------------------------------
__global__ void k_finalize(const float* __restrict__ z,
                           const float* __restrict__ cb,
                           const int* __restrict__ idx_ws,
                           float* __restrict__ zq_out,
                           float* __restrict__ loss_acc) {
    int n    = blockIdx.x * (blockDim.x >> 5) + (threadIdx.x >> 5);
    int lane = threadIdx.x & 31;
    if (n >= NPOS) return;
    int b = n >> 10, hw = n & (HW - 1);
    int ci = idx_ws[n];

    // normalize z[n] in f32
    const float* zbase = z + ((size_t)b * DIM) * HW + hw;
    float zv[8]; float zs = 0.f;
#pragma unroll
    for (int k = 0; k < 8; ++k) {
        float x = zbase[(size_t)(lane + 32 * k) * HW];
        zv[k] = x; zs += x * x;
    }
#pragma unroll
    for (int off = 16; off >= 1; off >>= 1) zs += __shfl_xor(zs, off, 32);
    float zscale = 1.0f / fmaxf(sqrtf(zs), 1e-12f);

    // normalize raw codebook row (== normalized codebook direction)
    const float* crow = cb + (size_t)ci * DIM;
    float cv[8]; float cs = 0.f;
#pragma unroll
    for (int k = 0; k < 8; ++k) {
        float x = crow[lane + 32 * k];
        cv[k] = x; cs += x * x;
    }
#pragma unroll
    for (int off = 16; off >= 1; off >>= 1) cs += __shfl_xor(cs, off, 32);
    float cscale = 1.0f / fmaxf(sqrtf(cs), 1e-12f);

    float lsum = 0.f;
#pragma unroll
    for (int k = 0; k < 8; ++k) {
        int d = lane + 32 * k;
        float zq = cv[k] * cscale;
        float zz = zv[k] * zscale;
        float df = zq - zz;
        lsum += df * df;
        zq_out[((size_t)b * DIM + d) * HW + hw] = zq;   // straight-through == z_q
    }
#pragma unroll
    for (int off = 16; off >= 1; off >>= 1) lsum += __shfl_xor(lsum, off, 32);
    if (lane == 0) atomicAdd(loss_acc, lsum);
}

// ---------------------------------------------------------------------------
// Kernel 5: loss = (1 + BETA) * mean((z_q - z)^2)
// ---------------------------------------------------------------------------
__global__ void k_loss(const float* __restrict__ loss_acc, float* __restrict__ out) {
    out[0] = (1.0f + BETA) * loss_acc[0] / ((float)NPOS * (float)DIM);
}

extern "C" void kernel_launch(void* const* d_in, const int* in_sizes, int n_in,
                              void* d_out, int out_size, void* d_ws, size_t ws_size,
                              hipStream_t stream) {
    const float* z  = (const float*)d_in[0];   // [16,256,32,32]
    const float* cb = (const float*)d_in[1];   // [8192,256]
    float* out = (float*)d_out;                // zq(4194304) | idx(16384) | loss(1)

    char* ws = (char*)d_ws;
    _Float16* e_norm = (_Float16*)(ws);                                   // 4 MB
    _Float16* z_norm = (_Float16*)(ws + (size_t)CB_K * DIM * 2);          // 8 MB
    int*      idx_ws = (int*)(ws + (size_t)CB_K * DIM * 2
                                 + (size_t)NPOS * DIM * 2);               // 64 KB
    float*    loss_acc = (float*)(ws + (size_t)CB_K * DIM * 2
                                     + (size_t)NPOS * DIM * 2
                                     + (size_t)NPOS * 4);

    float* zq_out  = out;
    float* idx_out = out + ZQ_ELEMS;
    float* loss_out = out + ZQ_ELEMS + NPOS;

    // 1. normalize codebook -> f16
    k_norm_codebook<<<CB_K / 8, 256, 0, stream>>>(cb, e_norm);
    // 2. normalize z -> f16 [N, 256]
    k_norm_z<<<NPOS / 32, 1024, 0, stream>>>(z, z_norm);
    // 3. WMMA similarity + argmax (128 blocks x 8 waves x 16 rows)
    k_argmax<<<NPOS / 128, 256, 0, stream>>>(z_norm, e_norm, idx_ws, idx_out);
    // 4. loss accumulator init + finalize
    k_zero<<<1, 1, 0, stream>>>(loss_acc);
    k_finalize<<<NPOS / 32, 1024, 0, stream>>>(z, cb, idx_ws, zq_out, loss_acc);
    // 5. scalar loss
    k_loss<<<1, 1, 0, stream>>>(loss_acc, loss_out);
}
